// JointPairHead_3272765079679
// MI455X (gfx1250) — compile-verified, matrix operand order checked by reading, not run on hardware
//
#include <hip/hip_runtime.h>
#include <hip/hip_bf16.h>

// ---------- constants ----------
#define D       256
#define NEDGE   262144
#define EPS     1e-5f

typedef __attribute__((ext_vector_type(16))) __bf16 bf16x16;
typedef __attribute__((ext_vector_type(8)))  __bf16 bf16x8;
typedef __attribute__((ext_vector_type(8)))  float  f32x8;

// ---- CDNA5 async global->LDS DMA (ASYNCcnt) availability ----
#if defined(__has_builtin)
#  if __has_builtin(__builtin_amdgcn_global_load_async_to_lds_b128) && \
      __has_builtin(__builtin_amdgcn_s_wait_asynccnt)
#    define HAVE_ASYNC_LDS 1
#  endif
#endif
#ifndef HAVE_ASYNC_LDS
#  define HAVE_ASYNC_LDS 0
#endif

#if HAVE_ASYNC_LDS
// builtin signature (from clang diagnostic): (int4 AS1*, int4 AS3*, int, int)
typedef int v4i __attribute__((ext_vector_type(4)));
typedef __attribute__((address_space(1))) v4i gv4i;   // global (printed __device__)
typedef __attribute__((address_space(3))) v4i lv4i;   // LDS    (printed __shared__)
// flat LDS address truncates to the AS(3) offset (ISA 10.2)
#define TO_GBL(p) ((gv4i*)(unsigned long long)(p))
#define TO_LDS(p) ((lv4i*)(unsigned)(unsigned long long)(p))
#define NBUF 2
#else
#define NBUF 1
#endif

// ---------------------------------------------------------------------------
// utility kernels
// ---------------------------------------------------------------------------
__global__ void zero_f32(float* p, int n) {
    int i = blockIdx.x * blockDim.x + threadIdx.x;
    if (i < n) p[i] = 0.0f;
}

__global__ void cvt_w_bf16(const float* __restrict__ W, __bf16* __restrict__ Wb, int n) {
    int i = blockIdx.x * blockDim.x + threadIdx.x;
    if (i < n) Wb[i] = (__bf16)W[i];
}

// ---------------------------------------------------------------------------
// Stage A: h = x[src] + x[dst]; LayerNorm(feature dim) * ln_w; -> bf16
// one wave32 per edge, 8 channels per lane
// ---------------------------------------------------------------------------
__global__ void edge_ln_kernel(const float* __restrict__ x,
                               const int*   __restrict__ ei,   // [2, NEDGE]
                               const float* __restrict__ lnw,  // [256]
                               __bf16*      __restrict__ out)  // [NEDGE, 256] bf16
{
    const int wave = threadIdx.x >> 5;
    const int lane = threadIdx.x & 31;
    const int e    = blockIdx.x * 8 + wave;

    const int s = ei[e];
    const int d = ei[NEDGE + e];
    const float4* xs = (const float4*)(x + (size_t)s * D);
    const float4* xd = (const float4*)(x + (size_t)d * D);

    const int c0 = lane * 8;
    float4 a0 = xs[lane * 2 + 0], a1 = xs[lane * 2 + 1];
    float4 b0 = xd[lane * 2 + 0], b1 = xd[lane * 2 + 1];

    float h[8];
    h[0] = a0.x + b0.x; h[1] = a0.y + b0.y; h[2] = a0.z + b0.z; h[3] = a0.w + b0.w;
    h[4] = a1.x + b1.x; h[5] = a1.y + b1.y; h[6] = a1.z + b1.z; h[7] = a1.w + b1.w;

    float s1 = 0.0f, s2 = 0.0f;
#pragma unroll
    for (int j = 0; j < 8; ++j) { s1 += h[j]; s2 += h[j] * h[j]; }
#pragma unroll
    for (int off = 16; off > 0; off >>= 1) {
        s1 += __shfl_xor(s1, off, 32);
        s2 += __shfl_xor(s2, off, 32);
    }
    const float mu  = s1 * (1.0f / D);
    const float var = s2 * (1.0f / D) - mu * mu;
    const float rs  = rsqrtf(var + EPS);

    bf16x8 o;
#pragma unroll
    for (int j = 0; j < 8; ++j)
        o[j] = (__bf16)((h[j] - mu) * rs * lnw[c0 + j]);
    *(bf16x8*)(out + (size_t)e * D + c0) = o;
}

// ---------------------------------------------------------------------------
// Fused GEMM + BatchNorm-stats layer.
//   C[e,o] = sum_k A'[e,k] * W[o,k]
//   MODE==0: A' = A (bf16, already normalized)
//   MODE==1: A' = relu( bf16(A)*scale[k] + bias[k] )   (fused prev BatchNorm)
// Workgroup: 8 waves; wave = 16 edges x 256 channels. K tiled by 32.
// W slab double-buffered in LDS via GLOBAL_LOAD_ASYNC_TO_LDS (overlapped DMA),
// A fragments double-buffered in regs, B fragments software-pipelined.
// ---------------------------------------------------------------------------
template <int MODE>
__global__ __launch_bounds__(256, 2)
void gemm_bn_kernel(const __bf16* __restrict__ A,      // [NEDGE,256] bf16
                    const __bf16* __restrict__ Wb,     // [256,256] bf16 (layer)
                    const float*  __restrict__ scale,  // [256] (MODE==1)
                    const float*  __restrict__ bias,   // [256] (MODE==1)
                    __bf16*       __restrict__ C,      // [NEDGE,256] bf16 raw
                    float*        __restrict__ gstat)  // [512]: sum, sumsq
{
    constexpr int WS = 40;                       // LDS row stride in bf16 (80 B)
    __shared__ __bf16 Wt[NBUF][256 * WS];
    __shared__ float  s_stat[512];
    __shared__ float  s_sc[256], s_bi[256];

    const int t    = threadIdx.x;
    const int lane = t & 31;
    const int col  = lane & 15;                  // N column within 16x16 tile
    const int sel  = lane >> 4;                  // K-half selector
    const int wave = t >> 5;
    const int band = blockIdx.x * 128 + wave * 16;

    s_stat[t]       = 0.0f;
    s_stat[256 + t] = 0.0f;
    if (MODE) { s_sc[t] = scale[t]; s_bi[t] = bias[t]; }

    f32x8 acc[16];
#pragma unroll
    for (int nt = 0; nt < 16; ++nt)
#pragma unroll
        for (int r = 0; r < 8; ++r) acc[nt][r] = 0.0f;

    const __bf16* Arow = A + (size_t)(band + col) * D;

    // A fragment double-buffer: issue ks=0 loads now
    bf16x8 ra = *(const bf16x8*)(Arow + 8 * sel);
    bf16x8 rb = *(const bf16x8*)(Arow + 16 + 8 * sel);

#if HAVE_ASYNC_LDS
    {   // DMA W slab 0 into buffer 0 (thread t owns W row t, 64 B)
        const __bf16* src = Wb + t * D;
        __bf16*       dst = &Wt[0][t * WS];
#pragma unroll
        for (int q = 0; q < 4; ++q)
            __builtin_amdgcn_global_load_async_to_lds_b128(
                TO_GBL(src + q * 8), TO_LDS(dst + q * 8), 0, 0);
        __builtin_amdgcn_s_wait_asynccnt(0);
    }
    __syncthreads();
#endif

    for (int ks = 0; ks < 8; ++ks) {
#if HAVE_ASYNC_LDS
        const int cur = ks & 1;
        if (ks < 7) {   // overlap next W slab DMA with this k-step's WMMAs
            const __bf16* src = Wb + t * D + (ks + 1) * 32;
            __bf16*       dst = &Wt[cur ^ 1][t * WS];
#pragma unroll
            for (int q = 0; q < 4; ++q)
                __builtin_amdgcn_global_load_async_to_lds_b128(
                    TO_GBL(src + q * 8), TO_LDS(dst + q * 8), 0, 0);
        }
#else
        const int cur = 0;
        __syncthreads();  // previous slab fully consumed
        {   // synchronous cooperative W slab load
            const bf16x8* wsrc = (const bf16x8*)(Wb + t * D + ks * 32);
            bf16x8*       wdst = (bf16x8*)(&Wt[0][t * WS]);
            wdst[0] = wsrc[0]; wdst[1] = wsrc[1];
            wdst[2] = wsrc[2]; wdst[3] = wsrc[3];
        }
        __syncthreads();
#endif
        // prefetch A fragment for next k-step (in flight during wmma burst)
        bf16x8 ran = ra, rbn = rb;
        if (ks < 7) {
            ran = *(const bf16x8*)(Arow + (ks + 1) * 32 + 8 * sel);
            rbn = *(const bf16x8*)(Arow + (ks + 1) * 32 + 16 + 8 * sel);
        }

        // Build A fragment (documented 16-bit A layout):
        // elems 0..7 -> K = ks*32 + 8*sel + j ; elems 8..15 -> +16
        bf16x16 afrag;
        if (MODE) {
            const int ka = ks * 32 + 8 * sel;
            const int kb = ka + 16;
            float4 s0 = *(const float4*)(s_sc + ka), s1 = *(const float4*)(s_sc + ka + 4);
            float4 s2 = *(const float4*)(s_sc + kb), s3 = *(const float4*)(s_sc + kb + 4);
            float4 b0 = *(const float4*)(s_bi + ka), b1 = *(const float4*)(s_bi + ka + 4);
            float4 b2 = *(const float4*)(s_bi + kb), b3 = *(const float4*)(s_bi + kb + 4);
            float sc[16] = {s0.x,s0.y,s0.z,s0.w, s1.x,s1.y,s1.z,s1.w,
                            s2.x,s2.y,s2.z,s2.w, s3.x,s3.y,s3.z,s3.w};
            float bi[16] = {b0.x,b0.y,b0.z,b0.w, b1.x,b1.y,b1.z,b1.w,
                            b2.x,b2.y,b2.z,b2.w, b3.x,b3.y,b3.z,b3.w};
#pragma unroll
            for (int j = 0; j < 8; ++j) {
                float f = (float)ra[j] * sc[j]     + bi[j];
                float g = (float)rb[j] * sc[8 + j] + bi[8 + j];
                afrag[j]     = (__bf16)fmaxf(f, 0.0f);
                afrag[8 + j] = (__bf16)fmaxf(g, 0.0f);
            }
        } else {
#pragma unroll
            for (int j = 0; j < 8; ++j) { afrag[j] = ra[j]; afrag[8 + j] = rb[j]; }
        }

        // software-pipelined B fragments: prefetch tile nt+1 during wmma nt
        const __bf16* wbase = &Wt[cur][0];
        bf16x8 blo = *(const bf16x8*)(wbase + col * WS + 8 * sel);
        bf16x8 bhi = *(const bf16x8*)(wbase + col * WS + 16 + 8 * sel);
#pragma unroll
        for (int nt = 0; nt < 16; ++nt) {
            bf16x8 nlo = blo, nhi = bhi;
            if (nt < 15) {
                const __bf16* wr = wbase + ((nt + 1) * 16 + col) * WS;
                nlo = *(const bf16x8*)(wr + 8 * sel);
                nhi = *(const bf16x8*)(wr + 16 + 8 * sel);
            }
            bf16x16 bfrag;
#pragma unroll
            for (int j = 0; j < 8; ++j) { bfrag[j] = blo[j]; bfrag[8 + j] = bhi[j]; }
            acc[nt] = __builtin_amdgcn_wmma_f32_16x16x32_bf16(
                false, afrag, false, bfrag, (short)0, acc[nt], false, false);
            blo = nlo; bhi = nhi;
        }

        ra = ran; rb = rbn;

#if HAVE_ASYNC_LDS
        if (ks < 7) __builtin_amdgcn_s_wait_asynccnt(0);
        __syncthreads();
#endif
    }

    // epilogue: bf16 store + per-channel stats
#pragma unroll
    for (int nt = 0; nt < 16; ++nt) {
        const int ch = nt * 16 + col;
        float ps = 0.0f, pq = 0.0f;
#pragma unroll
        for (int r = 0; r < 8; ++r) {
            float v = acc[nt][r];                       // row M = r + 8*sel
            ps += v; pq += v * v;
            C[(size_t)(band + r + 8 * sel) * D + ch] = (__bf16)v;
        }
        ps += __shfl_xor(ps, 16, 32);                   // lanes L,L+16 share channel
        pq += __shfl_xor(pq, 16, 32);
        if (sel == 0) {
            atomicAdd(&s_stat[ch], ps);                 // ds_add_f32
            atomicAdd(&s_stat[256 + ch], pq);
        }
    }
    __syncthreads();
    atomicAdd(&gstat[t],       s_stat[t]);              // global_atomic_add_f32
    atomicAdd(&gstat[256 + t], s_stat[256 + t]);
}

// ---------------------------------------------------------------------------
// Fold sums into per-channel scale/bias
// ---------------------------------------------------------------------------
__global__ void finalize_bn_kernel(const float* __restrict__ gstat,
                                   const float* __restrict__ gamma,
                                   const float* __restrict__ beta,
                                   float* __restrict__ sc, float* __restrict__ bi)
{
    const int c = threadIdx.x;
    const float mean = gstat[c] * (1.0f / NEDGE);
    const float var  = gstat[256 + c] * (1.0f / NEDGE) - mean * mean;
    const float s    = gamma[c] * rsqrtf(var + EPS);
    sc[c] = s;
    bi[c] = beta[c] - mean * s;
}

// ---------------------------------------------------------------------------
// logits[e] = sum_c relu(C2[e,c]*scale[c]+bias[c]) * Wout[c] ; wave per edge
// ---------------------------------------------------------------------------
__global__ void logits_kernel(const __bf16* __restrict__ C2,
                              const float*  __restrict__ sc,
                              const float*  __restrict__ bi,
                              const float*  __restrict__ wout,
                              float*        __restrict__ out)
{
    const int wave = threadIdx.x >> 5;
    const int lane = threadIdx.x & 31;
    const int e    = blockIdx.x * 8 + wave;
    const int c0   = lane * 8;

    bf16x8 v = *(const bf16x8*)(C2 + (size_t)e * D + c0);
    float acc = 0.0f;
#pragma unroll
    for (int j = 0; j < 8; ++j) {
        float f = (float)v[j] * sc[c0 + j] + bi[c0 + j];
        acc += fmaxf(f, 0.0f) * wout[c0 + j];
    }
#pragma unroll
    for (int off = 16; off > 0; off >>= 1) acc += __shfl_xor(acc, off, 32);
    if (lane == 0) out[e] = acc;
}

// ---------------------------------------------------------------------------
// launch
// ---------------------------------------------------------------------------
extern "C" void kernel_launch(void* const* d_in, const int* in_sizes, int n_in,
                              void* d_out, int out_size, void* d_ws, size_t ws_size,
                              hipStream_t stream)
{
    const float* x      = (const float*)d_in[0];   // [50000,256]
    const int*   ei     = (const int*)  d_in[1];   // [2,262144]
    const float* lnw    = (const float*)d_in[2];   // [256]
    const float* Ws     = (const float*)d_in[3];   // [3,256,256]
    const float* gammas = (const float*)d_in[4];   // [3,256]
    const float* betas  = (const float*)d_in[5];   // [3,256]
    const float* wout   = (const float*)d_in[6];   // [1,256]
    float*       out    = (float*)d_out;           // [262144]

    char* ws = (char*)d_ws;
    const size_t ACT = (size_t)NEDGE * D * sizeof(__bf16);   // 128 MB
    __bf16* bufA  = (__bf16*)ws;
    __bf16* bufB  = (__bf16*)(ws + ACT);
    __bf16* Wb    = (__bf16*)(ws + 2 * ACT);                 // 3*65536 bf16
    float*  gstat = (float*) (ws + 2 * ACT + 3 * 65536 * sizeof(__bf16)); // 3*512
    float*  scbi  = gstat + 3 * 512;                         // 3*512

    zero_f32<<<6, 256, 0, stream>>>(gstat, 3 * 512);
    cvt_w_bf16<<<768, 256, 0, stream>>>(Ws, Wb, 3 * 256 * 256);

    edge_ln_kernel<<<NEDGE / 8, 256, 0, stream>>>(x, ei, lnw, bufA);

    // layer 0
    gemm_bn_kernel<0><<<NEDGE / 128, 256, 0, stream>>>(bufA, Wb, nullptr, nullptr,
                                                       bufB, gstat);
    finalize_bn_kernel<<<1, 256, 0, stream>>>(gstat, gammas, betas,
                                              scbi, scbi + 256);
    // layer 1
    gemm_bn_kernel<1><<<NEDGE / 128, 256, 0, stream>>>(bufB, Wb + 65536,
                                                       scbi, scbi + 256,
                                                       bufA, gstat + 512);
    finalize_bn_kernel<<<1, 256, 0, stream>>>(gstat + 512, gammas + 256, betas + 256,
                                              scbi + 512, scbi + 768);
    // layer 2
    gemm_bn_kernel<1><<<NEDGE / 128, 256, 0, stream>>>(bufA, Wb + 131072,
                                                       scbi + 512, scbi + 768,
                                                       bufB, gstat + 1024);
    finalize_bn_kernel<<<1, 256, 0, stream>>>(gstat + 1024, gammas + 512, betas + 512,
                                              scbi + 1024, scbi + 1280);

    logits_kernel<<<NEDGE / 8, 256, 0, stream>>>(bufB, scbi + 1024, scbi + 1280,
                                                 wout, out);
}
